// NodeLevelAttention_55697135895016
// MI455X (gfx1250) — matrix operand order repaired and checked by reading, not compiled, not run on hardware
//
#include <hip/hip_runtime.h>
#include <hip/hip_bf16.h>

#define N_NODES 40000
#define N_EDGES 640000
#define IN_DIM  128
#define EDGE_DIM 64
#define HID     128

typedef __bf16 bf16;
typedef __attribute__((ext_vector_type(16))) __bf16 v16bf;
typedef __attribute__((ext_vector_type(8)))  __bf16 v8bf;
typedef __attribute__((ext_vector_type(4)))  __bf16 v4bf;
typedef __attribute__((ext_vector_type(8)))  float  v8f;
typedef __attribute__((ext_vector_type(4)))  float  v4f;

#if defined(__has_builtin)
#if __has_builtin(__builtin_amdgcn_sched_barrier)
#define SCHED_FENCE() __builtin_amdgcn_sched_barrier(0)
#endif
#endif
#ifndef SCHED_FENCE
#define SCHED_FENCE()
#endif

static __device__ __forceinline__ v8f wmma_bf16(v16bf a, v16bf b, v8f c) {
  // (neg_a, A, neg_b, B, c_mod, C, reuse_a, reuse_b)
  return __builtin_amdgcn_wmma_f32_16x16x32_bf16(false, a, false, b, (short)0, c,
                                                 false, false);
}

// Branch-free tanh: 1 - 2/(exp(2x)+1).  exp->inf gives 1, exp->0 gives -1.
static __device__ __forceinline__ float fast_tanh(float x) {
  float t = __expf(2.0f * x);
  return 1.0f - 2.0f / (t + 1.0f);
}

// A fragment (16x32 bf16). Layout: lanes 0-15 row M=lane, K in kOff..kOff+15;
// lanes 16-31 same row (lane&15), K in kOff+16..kOff+31.
// rowptr = base of THIS lane's row.
static __device__ __forceinline__ v16bf load_a_frag(const bf16* rowptr, int kOff) {
  int lane = threadIdx.x & 31;
  int kb = kOff + ((lane >> 4) << 4);
  v8bf lo = *reinterpret_cast<const v8bf*>(rowptr + kb);
  v8bf hi = *reinterpret_cast<const v8bf*>(rowptr + kb + 8);
  return __builtin_shufflevector(lo, hi, 0,1,2,3,4,5,6,7,8,9,10,11,12,13,14,15);
}

// B fragment (32x16 bf16) from K-transposed weights WT[n][k] (leading dim ldk):
// lane holds column n = nt*16 + (lane&15), K = kt*32 + (lane>>4)*16 .. +15
// -> one contiguous 32-byte load per lane.
static __device__ __forceinline__ v16bf load_b_frag(const bf16* WT, int ldk,
                                                    int nt, int kt) {
  int lane = threadIdx.x & 31;
  int n  = (nt << 4) + (lane & 15);
  int kb = (kt << 5) + ((lane >> 4) << 4);
  return *reinterpret_cast<const v16bf*>(WT + (size_t)n * ldk + kb);
}

static __device__ __forceinline__ void cvt_row_f32_to_bf16(const float* p, bf16* q,
                                                           int count) {
#pragma unroll
  for (int i = 0; i < count; i += 4) {
    v4f v = *reinterpret_cast<const v4f*>(p + i);
    v4bf o;
    o[0] = (bf16)v[0]; o[1] = (bf16)v[1]; o[2] = (bf16)v[2]; o[3] = (bf16)v[3];
    *reinterpret_cast<v4bf*>(q + i) = o;
  }
}

// ---------------- weight prep: bf16 K-transposed copies ----------------
__global__ void __launch_bounds__(256) k_prep(const float* __restrict__ Wn,
                                              const float* __restrict__ We,
                                              const float* __restrict__ Wa1,
                                              const float* __restrict__ Wo,
                                              bf16* __restrict__ WnT,
                                              bf16* __restrict__ WeT,
                                              bf16* __restrict__ Wa1T,
                                              bf16* __restrict__ WoT) {
  int tid = blockIdx.x * 256 + threadIdx.x;
  int stride = gridDim.x * 256;
  for (int i = tid; i < 128 * 128; i += stride) {
    int n = i & 127, k = i >> 7;
    WnT[n * 128 + k] = (bf16)Wn[k * 128 + n];
    WoT[n * 128 + k] = (bf16)Wo[k * 128 + n];
  }
  for (int i = tid; i < 128 * 64; i += stride) {
    int n = i & 127, k = i >> 7;
    WeT[n * 64 + k] = (bf16)We[k * 128 + n];
  }
  for (int i = tid; i < 128 * 384; i += stride) {
    int n = i & 127, k = i >> 7;
    Wa1T[n * 384 + k] = (bf16)Wa1[k * 128 + n];
  }
}

__global__ void __launch_bounds__(256) k_zero(float* __restrict__ agg,
                                              float* __restrict__ denom) {
  int tid = blockIdx.x * 256 + threadIdx.x;
  if (tid < N_NODES * HID) agg[tid] = 0.0f;
  if (tid < N_NODES) denom[tid] = 0.0f;
}

// ---------------- h = bf16(X @ W_node + b_node) ----------------
__global__ void __launch_bounds__(128) k_node_gemm(const float* __restrict__ X,
                                                   const bf16* __restrict__ WnT,
                                                   const float* __restrict__ bn,
                                                   bf16* __restrict__ h) {
  __shared__ bf16 sA[4 * 16 * 128];
  int wave = threadIdx.x >> 5;
  int lane = threadIdx.x & 31;
  int hf = lane >> 4;
  int r = lane & 15;
  int tile = blockIdx.x * 4 + wave;
  bf16* at = sA + wave * (16 * 128);

  cvt_row_f32_to_bf16(X + ((size_t)tile * 16 + r) * 128 + (hf << 6),
                      at + r * 128 + (hf << 6), 64);
  __syncthreads();

  v16bf a[4];
#pragma unroll
  for (int kt = 0; kt < 4; ++kt) a[kt] = load_a_frag(at + r * 128, kt * 32);

#pragma unroll
  for (int nt = 0; nt < 8; ++nt) {
    // preload the whole K-chain of B fragments, then fence the scheduler so the
    // loads all issue as one clause before the first WMMA (partial-wait overlap)
    v16bf b[4];
#pragma unroll
    for (int kt = 0; kt < 4; ++kt) b[kt] = load_b_frag(WnT, 128, nt, kt);
    float bv = bn[(nt << 4) + r];
    v8f c = {bv, bv, bv, bv, bv, bv, bv, bv};
    SCHED_FENCE();
#pragma unroll
    for (int kt = 0; kt < 4; ++kt) c = wmma_bf16(a[kt], b[kt], c);
    bf16* hp = h + (size_t)tile * 16 * 128 + (nt << 4) + r;
#pragma unroll
    for (int i = 0; i < 8; ++i)
      hp[(size_t)(i + (hf << 3)) * 128] = (bf16)c[i];
  }
}

// ---------------- edge pass 1: attention logits -> exp -> denom ----------------
__global__ void __launch_bounds__(128) k_edge_pass1(
    const float* __restrict__ Ef, const int* __restrict__ srcI,
    const int* __restrict__ dstI, const bf16* __restrict__ h,
    const bf16* __restrict__ WeT, const float* __restrict__ be,
    const bf16* __restrict__ Wa1T, const float* __restrict__ b1,
    const float* __restrict__ W2, const float* __restrict__ b2,
    float* __restrict__ attExp, float* __restrict__ denom) {
  __shared__ bf16 sEf[4 * 16 * 64];
  __shared__ bf16 sHe[4 * 16 * 128];
  int wave = threadIdx.x >> 5;
  int lane = threadIdx.x & 31;
  int hf = lane >> 4;
  int r = lane & 15;
  int tile = blockIdx.x * 4 + wave;
  int e0 = tile * 16;
  bf16* ef = sEf + wave * (16 * 64);
  bf16* he = sHe + wave * (16 * 128);

  // edge-feature tile f32 -> bf16 in LDS
  cvt_row_f32_to_bf16(Ef + ((size_t)e0 + r) * 64 + (hf << 5),
                      ef + r * 64 + (hf << 5), 32);
  __builtin_prefetch(srcI + e0 + 64, 0, 1);   // global_prefetch next tiles' indices
  __builtin_prefetch(dstI + e0 + 64, 0, 1);
  __syncthreads();

  // he = Ef @ W_edge + b_edge  (store bf16 tile in LDS)
  {
    v16bf aE0 = load_a_frag(ef + r * 64, 0);
    v16bf aE1 = load_a_frag(ef + r * 64, 32);
#pragma unroll
    for (int nt = 0; nt < 8; ++nt) {
      v16bf b0 = load_b_frag(WeT, 64, nt, 0);
      v16bf b1f = load_b_frag(WeT, 64, nt, 1);
      float bv = be[(nt << 4) + r];
      v8f c = {bv, bv, bv, bv, bv, bv, bv, bv};
      SCHED_FENCE();
      c = wmma_bf16(aE0, b0, c);
      c = wmma_bf16(aE1, b1f, c);
#pragma unroll
      for (int i = 0; i < 8; ++i)
        he[(i + (hf << 3)) * 128 + (nt << 4) + r] = (bf16)c[i];
    }
  }
  __syncthreads();

  // gather h[src], h[dst] rows as A fragments (h table is L2-resident bf16)
  int eR = e0 + r;
  int sN = srcI[eR];
  int dN = dstI[eR];
  const bf16* ps = h + (size_t)sN * 128;
  const bf16* pd = h + (size_t)dN * 128;
  v16bf aAll[12];                  // [0..3]=h_src, [4..7]=h_dst, [8..11]=he
#pragma unroll
  for (int kt = 0; kt < 4; ++kt) {
    aAll[kt]     = load_a_frag(ps, kt * 32);
    aAll[kt + 4] = load_a_frag(pd, kt * 32);
    aAll[kt + 8] = load_a_frag(he + r * 128, kt * 32);
  }

  // preload per-column bias / W_att2 values for all 8 column tiles
  float biasN[8], w2N[8];
#pragma unroll
  for (int nt = 0; nt < 8; ++nt) {
    biasN[nt] = b1[(nt << 4) + r];
    w2N[nt]   = W2[(nt << 4) + r];
  }
  SCHED_FENCE();   // issue all gathers + scalar preloads as one batch

  // T = tanh([h_src|h_dst|he] @ W_att1 + b1);  att = T @ W_att2
  float rowAcc[8] = {0, 0, 0, 0, 0, 0, 0, 0};
#pragma unroll
  for (int nt = 0; nt < 8; ++nt) {
    // preload the full 12-fragment B chain (24 b128 loads in one clause), then
    // fence so the WMMA chain overlaps the in-flight loads via partial waits
    v16bf b[12];
#pragma unroll
    for (int kt = 0; kt < 12; ++kt) b[kt] = load_b_frag(Wa1T, 384, nt, kt);
    float bv = biasN[nt];
    v8f c = {bv, bv, bv, bv, bv, bv, bv, bv};
    SCHED_FENCE();
#pragma unroll
    for (int kt = 0; kt < 12; ++kt) c = wmma_bf16(aAll[kt], b[kt], c);
    float w2v = w2N[nt];
#pragma unroll
    for (int i = 0; i < 8; ++i) rowAcc[i] += fast_tanh(c[i]) * w2v;
  }
  // reduce the 16 column-partials within each 16-lane half
#pragma unroll
  for (int m = 1; m <= 8; m <<= 1)
#pragma unroll
    for (int i = 0; i < 8; ++i) rowAcc[i] += __shfl_xor(rowAcc[i], m, 32);

  if (r < 8) {
    int row = (hf << 3) + r;     // lanes 0-7 -> rows 0-7, lanes 16-23 -> rows 8-15
    float ex = __expf(rowAcc[r] + b2[0]);   // unstabilized exp, matches reference
    int ee = e0 + row;
    attExp[ee] = ex;
    __hip_atomic_fetch_add(&denom[srcI[ee]], ex, __ATOMIC_RELAXED,
                           __HIP_MEMORY_SCOPE_AGENT);
  }
}

// ---------------- edge pass 2: agg[src] += h[dst] * coeff ----------------
__global__ void __launch_bounds__(256) k_edge_scatter(
    const int* __restrict__ srcI, const int* __restrict__ dstI,
    const bf16* __restrict__ h, const float* __restrict__ attExp,
    const float* __restrict__ denom, float* __restrict__ agg) {
  int gid = blockIdx.x * 256 + threadIdx.x;
  int e = gid >> 5;                 // one wave per edge
  int lane = threadIdx.x & 31;
  int s = srcI[e], d = dstI[e];
  float coeff = attExp[e] / denom[s];
  v4bf hv = *reinterpret_cast<const v4bf*>(h + (size_t)d * 128 + lane * 4);
  float* pa = agg + (size_t)s * 128 + lane * 4;
#pragma unroll
  for (int i = 0; i < 4; ++i)
    __hip_atomic_fetch_add(pa + i, (float)hv[i] * coeff, __ATOMIC_RELAXED,
                           __HIP_MEMORY_SCOPE_AGENT);
}

// ---------------- out = relu(LN(agg @ W_out + b_out)) ----------------
__global__ void __launch_bounds__(128) k_out_ln(const float* __restrict__ agg,
                                                const bf16* __restrict__ WoT,
                                                const float* __restrict__ bo,
                                                const float* __restrict__ gamma,
                                                const float* __restrict__ beta,
                                                float* __restrict__ out) {
  __shared__ bf16 sA[4 * 16 * 128];
  int wave = threadIdx.x >> 5;
  int lane = threadIdx.x & 31;
  int hf = lane >> 4;
  int r = lane & 15;
  int tile = blockIdx.x * 4 + wave;
  bf16* at = sA + wave * (16 * 128);

  cvt_row_f32_to_bf16(agg + ((size_t)tile * 16 + r) * 128 + (hf << 6),
                      at + r * 128 + (hf << 6), 64);
  __syncthreads();

  v16bf a[4];
#pragma unroll
  for (int kt = 0; kt < 4; ++kt) a[kt] = load_a_frag(at + r * 128, kt * 32);

  v8f acc[8];
#pragma unroll
  for (int nt = 0; nt < 8; ++nt) {
    v16bf b[4];
#pragma unroll
    for (int kt = 0; kt < 4; ++kt) b[kt] = load_b_frag(WoT, 128, nt, kt);
    float bv = bo[(nt << 4) + r];
    v8f c = {bv, bv, bv, bv, bv, bv, bv, bv};
    SCHED_FENCE();
#pragma unroll
    for (int kt = 0; kt < 4; ++kt) c = wmma_bf16(a[kt], b[kt], c);
    acc[nt] = c;
  }

  // per-row mean / var over 128 columns (8 fragments x 16 lanes per half)
  float sum[8] = {0, 0, 0, 0, 0, 0, 0, 0};
  float ssq[8] = {0, 0, 0, 0, 0, 0, 0, 0};
#pragma unroll
  for (int nt = 0; nt < 8; ++nt)
#pragma unroll
    for (int i = 0; i < 8; ++i) {
      float v = acc[nt][i];
      sum[i] += v;
      ssq[i] += v * v;
    }
#pragma unroll
  for (int m = 1; m <= 8; m <<= 1)
#pragma unroll
    for (int i = 0; i < 8; ++i) {
      sum[i] += __shfl_xor(sum[i], m, 32);
      ssq[i] += __shfl_xor(ssq[i], m, 32);
    }
  float mu[8], rs[8];
#pragma unroll
  for (int i = 0; i < 8; ++i) {
    mu[i] = sum[i] * (1.0f / 128.0f);
    float var = ssq[i] * (1.0f / 128.0f) - mu[i] * mu[i];
    rs[i] = rsqrtf(var + 1e-5f);
  }
#pragma unroll
  for (int nt = 0; nt < 8; ++nt) {
    int col = (nt << 4) + r;
    float g = gamma[col], bt = beta[col];
#pragma unroll
    for (int i = 0; i < 8; ++i) {
      int row = i + (hf << 3);
      float v = (acc[nt][i] - mu[i]) * rs[i] * g + bt;
      out[((size_t)tile * 16 + row) * 128 + col] = v > 0.0f ? v : 0.0f;
    }
  }
}

extern "C" void kernel_launch(void* const* d_in, const int* in_sizes, int n_in,
                              void* d_out, int out_size, void* d_ws, size_t ws_size,
                              hipStream_t stream) {
  const float* nodeF  = (const float*)d_in[0];
  const int*   eidx   = (const int*)d_in[1];
  const float* edgeF  = (const float*)d_in[2];
  const float* W_node = (const float*)d_in[3];
  const float* b_node = (const float*)d_in[4];
  const float* W_edge = (const float*)d_in[5];
  const float* b_edge = (const float*)d_in[6];
  const float* W_att1 = (const float*)d_in[7];
  const float* b_att1 = (const float*)d_in[8];
  const float* W_att2 = (const float*)d_in[9];
  const float* b_att2 = (const float*)d_in[10];
  const float* W_out  = (const float*)d_in[11];
  const float* b_out  = (const float*)d_in[12];
  const float* gammaP = (const float*)d_in[13];
  const float* betaP  = (const float*)d_in[14];
  float* out = (float*)d_out;

  const int* srcI = eidx;
  const int* dstI = eidx + N_EDGES;

  auto al = [](size_t x) { return (x + 255) & ~(size_t)255; };
  char* w = (char*)d_ws;
  bf16*  h     = (bf16*)w;  w += al((size_t)N_NODES * HID * 2);
  bf16*  WnT   = (bf16*)w;  w += al((size_t)128 * 128 * 2);
  bf16*  WeT   = (bf16*)w;  w += al((size_t)128 * 64 * 2);
  bf16*  Wa1T  = (bf16*)w;  w += al((size_t)128 * 384 * 2);
  bf16*  WoT   = (bf16*)w;  w += al((size_t)128 * 128 * 2);
  float* attE  = (float*)w; w += al((size_t)N_EDGES * 4);
  float* denom = (float*)w; w += al((size_t)N_NODES * 4);
  float* agg   = (float*)w; w += al((size_t)N_NODES * HID * 4);

  k_prep<<<64, 256, 0, stream>>>(W_node, W_edge, W_att1, W_out, WnT, WeT, Wa1T, WoT);
  k_zero<<<(N_NODES * HID + 255) / 256, 256, 0, stream>>>(agg, denom);
  k_node_gemm<<<N_NODES / 64, 128, 0, stream>>>(nodeF, WnT, b_node, h);
  k_edge_pass1<<<N_EDGES / 64, 128, 0, stream>>>(edgeF, srcI, dstI, h, WeT, b_edge,
                                                 Wa1T, b_att1, W_att2, b_att2,
                                                 attE, denom);
  k_edge_scatter<<<N_EDGES / 8, 256, 0, stream>>>(srcI, dstI, h, attE, denom, agg);
  k_out_ln<<<N_NODES / 64, 128, 0, stream>>>(agg, WoT, b_out, gammaP, betaP, out);
}
